// DMLSTM_88570815578783
// MI455X (gfx1250) — compile-verified, matrix operand order checked by reading, not run on hardware
//
#include <hip/hip_runtime.h>
#include <stdint.h>

typedef __attribute__((ext_vector_type(16))) __bf16 v16bf;
typedef __attribute__((ext_vector_type(8)))  float  v8f;

#define B_      64
#define S_      2048
#define I_      512
#define H_      512
#define N5_     2560      // 5*H
#define KSTEPS_ 32        // K=1024 / 32
#define KX_     16        // k-steps coming from x (I=512)
#define NTILES_ 160       // 2560/16

// round-to-nearest-even f32 -> bf16 (bit level; NaN edge ignored)
__device__ __forceinline__ uint16_t f32_bf16(float f) {
    uint32_t u = __float_as_uint(f);
    uint32_t r = u + 0x7FFFu + ((u >> 16) & 1u);
    return (uint16_t)(r >> 16);
}

// CDNA5 hardware transcendental tanh (v_tanh_f32).
__device__ __forceinline__ float fast_tanh(float x) {
#if __has_builtin(__builtin_amdgcn_tanhf)
    return __builtin_amdgcn_tanhf(x);
#else
    float y;
    asm volatile("v_tanh_f32 %0, %1\n\tv_nop" : "=v"(y) : "v"(x));
    return y;
#endif
}

// sigmoid(x) = 0.5*tanh(x/2) + 0.5  -> one hardware TRANS op + 2 VALU
__device__ __forceinline__ float fast_sigmoid(float x) {
    return __builtin_fmaf(0.5f, fast_tanh(0.5f * x), 0.5f);
}

// ---------------------------------------------------------------------------
// Pack W (f32, row-major [1024 x 2560]) into bf16 WMMA B-fragment layout:
// block bi = ntile*32 + kstep, 32 lanes x 16 bf16 (32B) per lane, contiguous.
// Per ISA 16-bit operand striping: lane<16 holds K {0..7,16..23}, lane>=16
// holds K {8..15,24..31}; N = ntile*16 + lane%16. For fixed (ntile,lane),
// consecutive k-steps sit 1024 B apart -> base + immediate-offset loads.
// ---------------------------------------------------------------------------
__global__ void pack_w(const float* __restrict__ W, uint16_t* __restrict__ Wbf) {
    int id   = blockIdx.x * blockDim.x + threadIdx.x;   // < 160*32*32*16
    int slot = id & 15;
    int lane = (id >> 4) & 31;
    int bi   = id >> 9;
    int kstep = bi & (KSTEPS_ - 1);
    int ntile = bi >> 5;
    int n  = ntile * 16 + (lane & 15);
    int hi = lane >> 4;
    int kr = (slot < 8) ? (slot + hi * 8) : (16 + (slot - 8) + hi * 8);
    int k  = kstep * 32 + kr;
    Wbf[id] = f32_bf16(W[k * N5_ + n]);
}

// ---------------------------------------------------------------------------
// Pack x_enc [B,S,I] f32 into per-t A-fragment bf16 layout:
// per t: 64 blocks (mtile*16 + kstep) x 32 lanes x 16 bf16 = 64KB.
// ---------------------------------------------------------------------------
__global__ void pack_x(const float* __restrict__ x, uint16_t* __restrict__ xbf) {
    long id  = (long)blockIdx.x * blockDim.x + threadIdx.x;  // < 2048*64*512
    int slot = (int)(id & 15);
    int lane = (int)((id >> 4) & 31);
    long blk = id >> 9;
    int bi   = (int)(blk & 63);
    int t    = (int)(blk >> 6);
    int kstep = bi & 15;
    int mtile = bi >> 4;
    int m  = mtile * 16 + (lane & 15);
    int hi = lane >> 4;
    int kr = (slot < 8) ? (slot + hi * 8) : (16 + (slot - 8) + hi * 8);
    int k  = kstep * 32 + kr;
    xbf[id] = f32_bf16(x[((long)m * S_ + t) * I_ + k]);
}

__global__ void init_state(float* __restrict__ c, uint16_t* __restrict__ h0) {
    int id = blockIdx.x * blockDim.x + threadIdx.x;
    if (id < B_ * H_) { c[id] = 0.0f; h0[id] = 0; }
}

// ---------------------------------------------------------------------------
// One recurrence step. Grid = 32 blocks (16 h-columns each, across ALL 5
// gates) x 128 threads (4 waves; wave w owns batch rows [16w,16w+16)).
// Register double-buffered k-loop: fragments for k+1 are loaded before the
// five WMMAs of step k are issued, keeping ~6 b128-pairs in flight so WMMAs
// don't drain loadcnt to 0. All loads are (hoisted base + immediate offset).
// Bias folds into the fused gate/highway epilogue (hardware v_tanh_f32).
// h is written back in bf16 fragment layout (double-buffered) for step t+1.
// ---------------------------------------------------------------------------
__global__ __launch_bounds__(128) void lstm_step(
    const uint16_t* __restrict__ xbf_t,   // A fragments (x part) for this t
    const uint16_t* __restrict__ Wbf,     // B fragments, all 160 n-tiles
    const float*    __restrict__ bias,    // [2560] f32
    const uint16_t* __restrict__ h_in,    // A fragments (h part), 64KB
    uint16_t*       __restrict__ h_out,   // next-step A fragments (h part)
    float*          __restrict__ c_state, // [64 x 512] f32
    float*          __restrict__ out,     // d_out base [B,S,H] f32
    int t)
{
    const int g     = blockIdx.x;         // h-chunk 0..31 (16 cols)
    const int hbase = g * 16;
    const int mtile = threadIdx.x >> 5;   // wave id 0..3
    const int lane  = threadIdx.x & 31;

    // loop-invariant fragment base pointers; per-k offsets are immediates
    const uint16_t* wb0 = Wbf + ((((0 * 32 + g) * KSTEPS_) * 32 + lane) << 4);
    const uint16_t* wb1 = Wbf + ((((1 * 32 + g) * KSTEPS_) * 32 + lane) << 4);
    const uint16_t* wb2 = Wbf + ((((2 * 32 + g) * KSTEPS_) * 32 + lane) << 4);
    const uint16_t* wb3 = Wbf + ((((3 * 32 + g) * KSTEPS_) * 32 + lane) << 4);
    const uint16_t* wb4 = Wbf + ((((4 * 32 + g) * KSTEPS_) * 32 + lane) << 4);
    const uint16_t* ax  = xbf_t + (((mtile * 16) * 32 + lane) << 4);
    const uint16_t* ah  = h_in  + (((mtile * 16) * 32 + lane) << 4);

    v8f acc[5];
#pragma unroll
    for (int gt = 0; gt < 5; ++gt)
#pragma unroll
        for (int i = 0; i < 8; ++i) acc[gt][i] = 0.0f;

    float bv[5];
#pragma unroll
    for (int gt = 0; gt < 5; ++gt)
        bv[gt] = bias[gt * H_ + hbase + (lane & 15)];

    // prime the pipeline: k=0 fragments
    v16bf a_cur = *(const v16bf*)ax;
    v16bf b_cur0 = *(const v16bf*)(wb0);
    v16bf b_cur1 = *(const v16bf*)(wb1);
    v16bf b_cur2 = *(const v16bf*)(wb2);
    v16bf b_cur3 = *(const v16bf*)(wb3);
    v16bf b_cur4 = *(const v16bf*)(wb4);

#pragma unroll
    for (int k = 0; k < KSTEPS_; ++k) {
        v16bf a_nxt, b_nxt0, b_nxt1, b_nxt2, b_nxt3, b_nxt4;
        if (k + 1 < KSTEPS_) {
            const int kn = k + 1;
            a_nxt = (kn < KX_) ? *(const v16bf*)(ax + kn * 512)
                               : *(const v16bf*)(ah + (kn - KX_) * 512);
            b_nxt0 = *(const v16bf*)(wb0 + kn * 512);
            b_nxt1 = *(const v16bf*)(wb1 + kn * 512);
            b_nxt2 = *(const v16bf*)(wb2 + kn * 512);
            b_nxt3 = *(const v16bf*)(wb3 + kn * 512);
            b_nxt4 = *(const v16bf*)(wb4 + kn * 512);
        }
        acc[0] = __builtin_amdgcn_wmma_f32_16x16x32_bf16(
            false, a_cur, false, b_cur0, (short)0, acc[0], false, false);
        acc[1] = __builtin_amdgcn_wmma_f32_16x16x32_bf16(
            false, a_cur, false, b_cur1, (short)0, acc[1], false, false);
        acc[2] = __builtin_amdgcn_wmma_f32_16x16x32_bf16(
            false, a_cur, false, b_cur2, (short)0, acc[2], false, false);
        acc[3] = __builtin_amdgcn_wmma_f32_16x16x32_bf16(
            false, a_cur, false, b_cur3, (short)0, acc[3], false, false);
        acc[4] = __builtin_amdgcn_wmma_f32_16x16x32_bf16(
            false, a_cur, false, b_cur4, (short)0, acc[4], false, false);
        if (k + 1 < KSTEPS_) {
            a_cur = a_nxt;
            b_cur0 = b_nxt0; b_cur1 = b_nxt1; b_cur2 = b_nxt2;
            b_cur3 = b_nxt3; b_cur4 = b_nxt4;
        }
    }

    // C/D layout: VGPR r, lane<16 -> (M=r, N=lane); lane>=16 -> (M=r+8, N=lane-16)
    const int hcol = hbase + (lane & 15);
    const int mhi  = (lane >> 4) * 8;
#pragma unroll
    for (int r = 0; r < 8; ++r) {
        int m = mtile * 16 + r + mhi;
        float zi = acc[0][r] + bv[0], zf = acc[1][r] + bv[1];
        float zo = acc[2][r] + bv[2], zc = acc[3][r] + bv[3];
        float zd = acc[4][r] + bv[4];
        float it = fast_sigmoid(zi);
        float ft = fast_sigmoid(zf);
        float ot = fast_sigmoid(zo);
        float c_old = c_state[m * H_ + hcol];
        float cp = ft * c_old + it * fast_tanh(zc);
        float dt = fast_tanh(zd);
        float cn = cp + dt * (cp - c_old);      // highway update
        float hn = ot * fast_tanh(cn);
        c_state[m * H_ + hcol] = cn;
        out[((long)m * S_ + t) * H_ + hcol] = hn;
        // scatter h into next-step A-fragment layout
        int kstep = hcol >> 5, kr = hcol & 31;
        int lane2 = (m & 15) + (((kr >> 3) & 1) << 4);
        int slot  = (kr & 7)  + (((kr >> 4) & 1) << 3);
        h_out[((((m >> 4) * 16 + kstep) * 32 + lane2) << 4) + slot] = f32_bf16(hn);
    }
}

extern "C" void kernel_launch(void* const* d_in, const int* in_sizes, int n_in,
                              void* d_out, int out_size, void* d_ws, size_t ws_size,
                              hipStream_t stream) {
    const float* x = (const float*)d_in[0];   // [64,2048,512]
    const float* W = (const float*)d_in[1];   // [1024,2560]
    const float* b = (const float*)d_in[2];   // [2560]
    float* out = (float*)d_out;

    // workspace layout (bytes):
    //   Wbf : 1024*2560*2            =   5,242,880
    //   xbf : 2048*64*512*2          = 134,217,728
    //   hbf : 2 x 64KB double buffer =     131,072
    //   c   : 64*512*4               =     131,072
    uint8_t* ws = (uint8_t*)d_ws;
    uint16_t* Wbf  = (uint16_t*)ws;
    uint16_t* xbf  = (uint16_t*)(ws + 5242880);
    uint16_t* hbf0 = (uint16_t*)(ws + 5242880 + 134217728);
    uint16_t* hbf1 = hbf0 + 32768;
    float*    cst  = (float*)(hbf1 + 32768);

    pack_w<<<(NTILES_ * KSTEPS_ * 32 * 16) / 256, 256, 0, stream>>>(W, Wbf);
    pack_x<<<(int)((long)S_ * 64 * 512 / 256), 256, 0, stream>>>(x, xbf);
    init_state<<<(B_ * H_ + 255) / 256, 256, 0, stream>>>(cst, hbf0);

    for (int t = 0; t < S_; ++t) {
        const uint16_t* hin  = (t & 1) ? hbf1 : hbf0;
        uint16_t*       hout = (t & 1) ? hbf0 : hbf1;
        lstm_step<<<32, 128, 0, stream>>>(xbf + (long)t * 32768, Wbf, b,
                                          hin, hout, cst, out, t);
    }
}
